// Model_36464272343406
// MI455X (gfx1250) — matrix operand order repaired and checked by reading, not compiled, not run on hardware
//
#include <hip/hip_runtime.h>

// ---------------- problem constants (baked from reference) ----------------
#define BB     2
#define NN     16
#define EXTC   64          // CHUNK + LAH
#define LAH    4
#define SS     (NN * EXTC) // 1024
#define TT     (BB * SS)   // 2048 tokens
#define DD     512
#define FFD    2048
#define HH     8
#define DHD    64
#define KW     31
#define LAYERS 6

#if defined(__HIP_DEVICE_COMPILE__)
typedef __attribute__((ext_vector_type(16))) __bf16 v16bf;
typedef __attribute__((ext_vector_type(8)))  __bf16 v8bf;
typedef __attribute__((ext_vector_type(8)))  float  v8f;

static __device__ __forceinline__ __bf16 f2bf(float f) {
  unsigned u = __builtin_bit_cast(unsigned, f);
  u += 0x7FFFu + ((u >> 16) & 1u);           // round-to-nearest-even
  unsigned short s = (unsigned short)(u >> 16);
  return __builtin_bit_cast(__bf16, s);
}

static __device__ __forceinline__ v16bf cat16(v8bf lo, v8bf hi) {
  return __builtin_shufflevector(lo, hi, 0, 1, 2, 3, 4, 5, 6, 7,
                                 8, 9, 10, 11, 12, 13, 14, 15);
}

#define WMMA_BF16(Afrag, Bfrag, Cfrag)                                        \
  __builtin_amdgcn_wmma_f32_16x16x32_bf16(false, (Afrag), false, (Bfrag),     \
                                          (short)0, (Cfrag), false, false)
#endif

// ---------------- weight transpose + f32->bf16 convert --------------------
// W [Kd, Nd] row-major f32  ->  Wt [Nd, Kd] row-major bf16
__global__ void wconvT(const float* __restrict__ W, unsigned short* __restrict__ Wt_,
                       int Kd, int Nd) {
#if defined(__HIP_DEVICE_COMPILE__)
  __bf16* Wt = (__bf16*)Wt_;
  int idx = blockIdx.x * 256 + threadIdx.x;
  int tot = Kd * Nd;
  if (idx >= tot) return;
  int k = idx / Nd;
  int n = idx - k * Nd;
  Wt[(size_t)n * Kd + k] = f2bf(W[idx]);
#endif
}

// ---------------- LayerNorm over D=512 (fused SiLU; f32 or bf16 out) ------
__global__ void ln_kernel(const float* __restrict__ in, const float* __restrict__ g,
                          const float* __restrict__ bta, void* __restrict__ out_,
                          int applySilu, int outBf16) {
#if defined(__HIP_DEVICE_COMPILE__)
  const int t   = blockIdx.x;
  const int tid = threadIdx.x;
  __shared__ float red[256];
  float x0 = in[(size_t)t * DD + tid];
  float x1 = in[(size_t)t * DD + 256 + tid];
  red[tid] = x0 + x1;
  __syncthreads();
  for (int o = 128; o > 0; o >>= 1) {
    if (tid < o) red[tid] += red[tid + o];
    __syncthreads();
  }
  float mean = red[0] * (1.0f / DD);
  __syncthreads();
  float d0 = x0 - mean, d1 = x1 - mean;
  red[tid] = d0 * d0 + d1 * d1;
  __syncthreads();
  for (int o = 128; o > 0; o >>= 1) {
    if (tid < o) red[tid] += red[tid + o];
    __syncthreads();
  }
  float rstd = rsqrtf(red[0] * (1.0f / DD) + 1e-5f);
  float y0 = d0 * rstd * g[tid] + bta[tid];
  float y1 = d1 * rstd * g[256 + tid] + bta[256 + tid];
  if (applySilu) {
    y0 = y0 / (1.f + __expf(-y0));
    y1 = y1 / (1.f + __expf(-y1));
  }
  if (outBf16) {
    __bf16* out = (__bf16*)out_;
    out[(size_t)t * DD + tid]       = f2bf(y0);
    out[(size_t)t * DD + 256 + tid] = f2bf(y1);
  } else {
    float* out = (float*)out_;
    out[(size_t)t * DD + tid]       = y0;
    out[(size_t)t * DD + 256 + tid] = y1;
  }
#endif
}

// ---------------- WMMA GEMM: C[T,N] = A_bf16[T,K] * Wt_bf16[N,K]^T + bias -
// mode 0: store f32 C    mode 1: store bf16 silu(C)   mode 2: resid += rscale*C
// grid = (Nout/64, TT/64), block = 128 (4 waves); each wave: 16x64 output.
__global__ void gemm_kernel(const unsigned short* __restrict__ A_,
                            const unsigned short* __restrict__ Wt_,
                            const float* __restrict__ bias, float* __restrict__ Cout,
                            unsigned short* __restrict__ CoutBf_,
                            float* __restrict__ resid, int Nout, int Kd, int mode,
                            float rscale) {
#if defined(__HIP_DEVICE_COMPILE__)
  const __bf16* A  = (const __bf16*)A_;
  const __bf16* Wt = (const __bf16*)Wt_;
  __bf16* CoutBf   = (__bf16*)CoutBf_;
  const int lane = threadIdx.x & 31;
  const int wid  = threadIdx.x >> 5;
  const int mt   = blockIdx.y * 4 + wid;   // 16-row tile index
  const int nb   = blockIdx.x * 64;        // 64-col strip base
  const int half = lane >> 4;
  const int ln   = lane & 15;
  const int row  = mt * 16 + ln;

  v8f acc[4] = {v8f{}, v8f{}, v8f{}, v8f{}};

  for (int k0 = 0; k0 < Kd; k0 += 32) {
    // A fragment (16x32 bf16): lanes 0-15 K {0..15}; lanes 16-31 K {16..31}
    const __bf16* ap = A + (size_t)row * Kd + k0 + half * 16;
    v16bf af = cat16(*(const v8bf*)(ap), *(const v8bf*)(ap + 8));

    const int kr = k0 + half * 16;   // B rows: lanes 0-15 -> k 0..15; 16-31 -> k 16..31
#pragma unroll
    for (int n = 0; n < 4; ++n) {
      const __bf16* wp = Wt + (size_t)(nb + n * 16 + ln) * Kd + kr;
      v16bf bf = *(const v16bf*)wp;
      __builtin_prefetch((const void*)(wp + 32), 0, 1);  // next K-step weight tile
      acc[n] = WMMA_BF16(af, bf, acc[n]);
    }
  }

#pragma unroll
  for (int n = 0; n < 4; ++n) {
    const int col = nb + n * 16 + ln;
    const float bv = bias[col];
#pragma unroll
    for (int r = 0; r < 8; ++r) {
      const int orow = mt * 16 + r + half * 8;
      float v = acc[n][r] + bv;
      if (mode == 2) {
        resid[(size_t)orow * Nout + col] += rscale * v;
      } else if (mode == 1) {
        v = v / (1.f + __expf(-v));
        CoutBf[(size_t)orow * Nout + col] = f2bf(v);
      } else {
        Cout[(size_t)orow * Nout + col] = v;
      }
    }
  }
#endif
}

// ---------------- QKV split: f32 [T,3D] -> bf16 Q,K [B,H,S,DH], Vt [B,H,DH,S]
__global__ void qkv_split(const float* __restrict__ qkv, unsigned short* __restrict__ qb_,
                          unsigned short* __restrict__ kb_, unsigned short* __restrict__ vt_) {
#if defined(__HIP_DEVICE_COMPILE__)
  __bf16* qb = (__bf16*)qb_;
  __bf16* kb = (__bf16*)kb_;
  __bf16* vt = (__bf16*)vt_;
  int idx = blockIdx.x * 256 + threadIdx.x;  // over T*D
  int t = idx >> 9;
  int d = idx & (DD - 1);
  int b = t >> 10;
  int s = t & (SS - 1);
  int h  = d >> 6;
  int dh = d & 63;
  size_t base = (size_t)t * (3 * DD);
  float q = qkv[base + d] * 0.125f;          // 1/sqrt(DH) pre-applied
  float k = qkv[base + DD + d];
  float v = qkv[base + 2 * DD + d];
  size_t o = ((size_t)(b * HH + h) * SS + s) * DHD + dh;
  qb[o] = f2bf(q);
  kb[o] = f2bf(k);
  vt[((size_t)(b * HH + h) * DHD + dh) * SS + s] = f2bf(v);
#endif
}

// ---------------- masked flash attention, one wave per (b,h,16-query tile)
__global__ void attn_kernel(const unsigned short* __restrict__ Qb_,
                            const unsigned short* __restrict__ Kb_,
                            const unsigned short* __restrict__ Vt_,
                            const unsigned char* __restrict__ seqmask,
                            unsigned short* __restrict__ attout_) {
#if defined(__HIP_DEVICE_COMPILE__)
  const __bf16* Qb = (const __bf16*)Qb_;
  const __bf16* Kb = (const __bf16*)Kb_;
  const __bf16* Vt = (const __bf16*)Vt_;
  __bf16* attout   = (__bf16*)attout_;
  const int qt   = blockIdx.x;     // 0..63 (16-query tiles)
  const int h    = blockIdx.y;     // 0..7
  const int b    = blockIdx.z;     // 0..1
  const int lane = threadIdx.x & 31;
  const int half = lane >> 4;
  const int ln   = lane & 15;
  const size_t qkoff = (size_t)(b * HH + h) * SS * DHD;
  const size_t voff  = (size_t)(b * HH + h) * DHD * SS;

  __shared__ __bf16 pl[16 * 32];

  // Q fragments for dims 0..31 and 32..63 (pre-scaled by 1/sqrt(DH))
  v16bf aq0, aq1;
  {
    const __bf16* qp = Qb + qkoff + (size_t)(qt * 16 + ln) * DHD + half * 16;
    aq0 = cat16(*(const v8bf*)(qp),      *(const v8bf*)(qp + 8));
    aq1 = cat16(*(const v8bf*)(qp + 32), *(const v8bf*)(qp + 40));
  }

  v8f acc[4] = {v8f{}, v8f{}, v8f{}, v8f{}};
  float m[8], l[8];
#pragma unroll
  for (int r = 0; r < 8; ++r) { m[r] = -3.0e38f; l[r] = 0.f; }

  const int qc  = qt >> 2;          // chunk index of this query tile
  const int nkb = (qc + 1) * 2;     // number of 32-key blocks to visit

  for (int kb32 = 0; kb32 < nkb; ++kb32) {
    const int keybase = kb32 * 32;
    v8f s[2];
#pragma unroll
    for (int t2 = 0; t2 < 2; ++t2) {
      const int key = keybase + t2 * 16 + ln;
      const __bf16* kp = Kb + qkoff + (size_t)key * DHD + half * 16;
      v16bf blo = *(const v16bf*)(kp);
      v16bf bhi = *(const v16bf*)(kp + 32);
      v8f sc = {};
      sc = WMMA_BF16(aq0, blo, sc);
      sc = WMMA_BF16(aq1, bhi, sc);
      // chunk mask: past chunks only real frames (w < EXT-LAH); current chunk full
      const int kc = key >> 6;
      const int w  = key & 63;
      const bool allow = ((kc < qc) && (w < EXTC - LAH)) || (kc == qc);
      const bool valid = seqmask[b * SS + key] != 0;
      if (!(allow && valid)) {
#pragma unroll
        for (int r = 0; r < 8; ++r) sc[r] = -1e30f;
      }
      s[t2] = sc;
    }

    // online softmax update (rows M = r + 8*half; reduce across 16 lanes of half)
    float fac[8];
#pragma unroll
    for (int r = 0; r < 8; ++r) {
      float tm = fmaxf(s[0][r], s[1][r]);
      for (int msk = 1; msk < 16; msk <<= 1) tm = fmaxf(tm, __shfl_xor(tm, msk, 32));
      float nm = fmaxf(m[r], tm);
      fac[r] = __expf(m[r] - nm);
      float p0 = __expf(s[0][r] - nm);
      float p1 = __expf(s[1][r] - nm);
      float ts = p0 + p1;
      for (int msk = 1; msk < 16; msk <<= 1) ts += __shfl_xor(ts, msk, 32);
      l[r] = l[r] * fac[r] + ts;
      m[r] = nm;
      pl[(r + half * 8) * 32 + ln]      = f2bf(p0);
      pl[(r + half * 8) * 32 + 16 + ln] = f2bf(p1);
    }
    __syncthreads();

    // reload P in A-fragment layout (two ds_load_b128 per lane)
    const __bf16* pp = pl + ln * 32 + half * 16;
    v16bf pf = cat16(*(const v8bf*)(pp), *(const v8bf*)(pp + 8));

#pragma unroll
    for (int n = 0; n < 4; ++n) {
      const __bf16* vp = Vt + voff + (size_t)(n * 16 + ln) * SS + keybase + half * 16;
      v16bf bv = *(const v16bf*)vp;
#pragma unroll
      for (int r = 0; r < 8; ++r) acc[n][r] *= fac[r];
      acc[n] = WMMA_BF16(pf, bv, acc[n]);
    }
    __syncthreads();
  }

#pragma unroll
  for (int r = 0; r < 8; ++r) l[r] = 1.f / l[r];
#pragma unroll
  for (int n = 0; n < 4; ++n) {
#pragma unroll
    for (int r = 0; r < 8; ++r) {
      const int qrow = qt * 16 + r + half * 8;
      attout[(size_t)(b * SS + qrow) * DD + h * DHD + n * 16 + ln] =
          f2bf(acc[n][r] * l[r]);
    }
  }
#endif
}

// ---------------- GLU: [T,2D] -> [T,D] ------------------------------------
__global__ void glu_kernel(const float* __restrict__ in2d, float* __restrict__ out) {
  int idx = blockIdx.x * 256 + threadIdx.x;   // over T*D
  int t = idx >> 9;
  int d = idx & (DD - 1);
  float a = in2d[(size_t)t * (2 * DD) + d];
  float g = in2d[(size_t)t * (2 * DD) + DD + d];
  out[idx] = a / (1.f + __expf(-g));
}

// ---------------- depthwise conv K=31 over flattened S --------------------
__global__ void dwconv_kernel(const float* __restrict__ in, const float* __restrict__ w,
                              const float* __restrict__ bias, float* __restrict__ out) {
  int idx = blockIdx.x * 256 + threadIdx.x;   // over T*D
  int t = idx >> 9;
  int d = idx & (DD - 1);
  int b = t >> 10;
  int s = t & (SS - 1);
  float sum = bias[d];
#pragma unroll
  for (int k = 0; k < KW; ++k) {
    int ss = s + k - KW / 2;
    if (ss >= 0 && ss < SS) sum += in[(size_t)(b * SS + ss) * DD + d] * w[k * DD + d];
  }
  out[idx] = sum;
}

// ---------------- zero padded frames --------------------------------------
__global__ void maskzero_kernel(float* __restrict__ hbuf, const unsigned char* __restrict__ m) {
  int idx = blockIdx.x * 256 + threadIdx.x;   // over T*D
  int t = idx >> 9;
  if (!m[t]) hbuf[idx] = 0.f;
}

// ---------------- strip lookahead frames: [B,N,60,D] ----------------------
__global__ void out_kernel(const float* __restrict__ hbuf, float* __restrict__ out, int total) {
  int idx = blockIdx.x * 256 + threadIdx.x;
  if (idx >= total) return;
  int d = idx & (DD - 1);
  int r = idx >> 9;
  int w = r % (EXTC - LAH); r /= (EXTC - LAH);
  int n = r & (NN - 1);
  int b = r >> 4;
  out[idx] = hbuf[((size_t)((b * NN + n) * EXTC + w) << 9) + d];
}

// ===========================================================================
extern "C" void kernel_launch(void* const* d_in, const int* in_sizes, int n_in,
                              void* d_out, int out_size, void* d_ws, size_t ws_size,
                              hipStream_t stream) {
  const float* x = (const float*)d_in[0];
  const unsigned char* smask = (const unsigned char*)d_in[1];  // bool -> 1 byte
  // (d_in[2] = lookahead_size, fixed at 4 in the compiled constants)
  const float* ff1_lng = (const float*)d_in[3];
  const float* ff1_lnb = (const float*)d_in[4];
  const float* ff1_w1  = (const float*)d_in[5];
  const float* ff1_b1  = (const float*)d_in[6];
  const float* ff1_w2  = (const float*)d_in[7];
  const float* ff1_b2  = (const float*)d_in[8];
  const float* att_lng = (const float*)d_in[9];
  const float* att_lnb = (const float*)d_in[10];
  const float* wqkv    = (const float*)d_in[11];
  const float* bqkv    = (const float*)d_in[12];
  const float* wo      = (const float*)d_in[13];
  const float* bo      = (const float*)d_in[14];
  const float* conv_lng= (const float*)d_in[15];
  const float* conv_lnb= (const float*)d_in[16];
  const float* pw1_w   = (const float*)d_in[17];
  const float* pw1_b   = (const float*)d_in[18];
  const float* dw_w    = (const float*)d_in[19];
  const float* dw_b    = (const float*)d_in[20];
  const float* cn_g    = (const float*)d_in[21];
  const float* cn_b    = (const float*)d_in[22];
  const float* pw2_w   = (const float*)d_in[23];
  const float* pw2_b   = (const float*)d_in[24];
  const float* ff2_lng = (const float*)d_in[25];
  const float* ff2_lnb = (const float*)d_in[26];
  const float* ff2_w1  = (const float*)d_in[27];
  const float* ff2_b1  = (const float*)d_in[28];
  const float* ff2_w2  = (const float*)d_in[29];
  const float* ff2_b2  = (const float*)d_in[30];
  const float* fin_g   = (const float*)d_in[31];
  const float* fin_b   = (const float*)d_in[32];

  // ---- workspace carve (≈48 MB) ----
  char* p = (char*)d_ws;
  auto carve = [&](size_t bytes) {
    void* r = (void*)p;
    p += (bytes + 255) & ~(size_t)255;
    return r;
  };
  float* h   = (float*)carve((size_t)TT * DD * 4);          // residual stream (f32)
  float* y   = (float*)carve((size_t)TT * DD * 4);          // f32 temp (glu out)
  float* c2  = (float*)carve((size_t)TT * DD * 4);          // f32 temp (dwconv out)
  float* big = (float*)carve((size_t)TT * FFD * 4);         // f32: qkv / pw1 out
  unsigned short* ybf  = (unsigned short*)carve((size_t)TT * DD * 2);   // LN out (bf16)
  unsigned short* midbf= (unsigned short*)carve((size_t)TT * FFD * 2);  // FF mid (bf16)
  unsigned short* abf  = (unsigned short*)carve((size_t)TT * DD * 2);   // attn out (bf16)
  unsigned short* qb   = (unsigned short*)carve((size_t)TT * DD * 2);
  unsigned short* kb   = (unsigned short*)carve((size_t)TT * DD * 2);
  unsigned short* vt   = (unsigned short*)carve((size_t)TT * DD * 2);
  unsigned short* wbuf = (unsigned short*)carve((size_t)DD * FFD * 2);
  (void)ws_size; (void)in_sizes; (void)n_in;

  hipMemcpyAsync(h, x, (size_t)TT * DD * 4, hipMemcpyDeviceToDevice, stream);

  auto WT = [&](const float* W, int Kd, int Nd) {
    int tot = Kd * Nd;
    wconvT<<<(tot + 255) / 256, 256, 0, stream>>>(W, wbuf, Kd, Nd);
  };
  auto GEMM = [&](const unsigned short* A, const float* bias, float* Cout,
                  unsigned short* CoutBf, float* resid, int Nout, int Kd, int mode,
                  float rscale) {
    gemm_kernel<<<dim3(Nout / 64, TT / 64), dim3(128), 0, stream>>>(
        A, wbuf, bias, Cout, CoutBf, resid, Nout, Kd, mode, rscale);
  };

  const int TD = TT * DD;
  for (int lay = 0; lay < LAYERS; ++lay) {
    // ---- FF1: h += 0.5 * (silu(ln(h) W1 + b1) W2 + b2)
    ln_kernel<<<TT, 256, 0, stream>>>(h, ff1_lng + lay * DD, ff1_lnb + lay * DD, ybf, 0, 1);
    WT(ff1_w1 + (size_t)lay * DD * FFD, DD, FFD);
    GEMM(ybf, ff1_b1 + lay * FFD, nullptr, midbf, nullptr, FFD, DD, 1, 0.f);
    WT(ff1_w2 + (size_t)lay * FFD * DD, FFD, DD);
    GEMM(midbf, ff1_b2 + lay * DD, nullptr, nullptr, h, DD, FFD, 2, 0.5f);

    // ---- MHSA with chunked mask
    ln_kernel<<<TT, 256, 0, stream>>>(h, att_lng + lay * DD, att_lnb + lay * DD, ybf, 0, 1);
    WT(wqkv + (size_t)lay * DD * 3 * DD, DD, 3 * DD);
    GEMM(ybf, bqkv + lay * 3 * DD, big, nullptr, nullptr, 3 * DD, DD, 0, 0.f);
    qkv_split<<<TD / 256, 256, 0, stream>>>(big, qb, kb, vt);
    attn_kernel<<<dim3(SS / 16, HH, BB), 32, 0, stream>>>(qb, kb, vt, smask, abf);
    WT(wo + (size_t)lay * DD * DD, DD, DD);
    GEMM(abf, bo + lay * DD, nullptr, nullptr, h, DD, DD, 2, 1.0f);
    maskzero_kernel<<<TD / 256, 256, 0, stream>>>(h, smask);

    // ---- conv module
    ln_kernel<<<TT, 256, 0, stream>>>(h, conv_lng + lay * DD, conv_lnb + lay * DD, ybf, 0, 1);
    WT(pw1_w + (size_t)lay * DD * 2 * DD, DD, 2 * DD);
    GEMM(ybf, pw1_b + lay * 2 * DD, big, nullptr, nullptr, 2 * DD, DD, 0, 0.f);
    glu_kernel<<<TD / 256, 256, 0, stream>>>(big, y);
    dwconv_kernel<<<TD / 256, 256, 0, stream>>>(y, dw_w + (size_t)lay * KW * DD,
                                                dw_b + lay * DD, c2);
    ln_kernel<<<TT, 256, 0, stream>>>(c2, cn_g + lay * DD, cn_b + lay * DD, ybf, 1, 1);
    WT(pw2_w + (size_t)lay * DD * DD, DD, DD);
    GEMM(ybf, pw2_b + lay * DD, nullptr, nullptr, h, DD, DD, 2, 1.0f);

    // ---- FF2
    ln_kernel<<<TT, 256, 0, stream>>>(h, ff2_lng + lay * DD, ff2_lnb + lay * DD, ybf, 0, 1);
    WT(ff2_w1 + (size_t)lay * DD * FFD, DD, FFD);
    GEMM(ybf, ff2_b1 + lay * FFD, nullptr, midbf, nullptr, FFD, DD, 1, 0.f);
    WT(ff2_w2 + (size_t)lay * FFD * DD, FFD, DD);
    GEMM(midbf, ff2_b2 + lay * DD, nullptr, nullptr, h, DD, FFD, 2, 0.5f);

    // ---- final per-layer LN (in place, f32)
    ln_kernel<<<TT, 256, 0, stream>>>(h, fin_g + lay * DD, fin_b + lay * DD, h, 0, 0);
  }

  out_kernel<<<(out_size + 255) / 256, 256, 0, stream>>>(h, (float*)d_out, out_size);
}
